// MiniJMamba_64561948393809
// MI455X (gfx1250) — matrix operand order repaired and checked by reading, not compiled, not run on hardware
//
#include <hip/hip_runtime.h>
#include <hip/hip_bf16.h>
#include <stdint.h>

// ---------------------------------------------------------------------------
// MiniJMamba forward for gfx1250 (MI455X).
//  - GEMMs: v_wmma_f32_16x16x32_f16, 128x128 block tile, 8 waves,
//    8 WMMAs/wave/K-step. Double-buffered LDS: tile i+1 staged with
//    GLOBAL_LOAD_ASYNC_TO_LDS_B128 (ASYNCcnt) + register-transposed W tile
//    while tile i is consumed by WMMAs.
//  - Attention: flash-style streaming softmax (no SxS scores).
// ---------------------------------------------------------------------------

typedef __attribute__((ext_vector_type(16))) _Float16 v16h;
typedef __attribute__((ext_vector_type(8)))  _Float16 v8h;
typedef __attribute__((ext_vector_type(8)))  float    v8f;

#define MROWS 8192      // B*S
#define SEQ   2048
#define NB    4
#define NH    8
#define HDIM  64
#define DMODEL 512

// ------------------------------ elementwise --------------------------------

__global__ void k_f32_to_f16(const float* __restrict__ s, _Float16* __restrict__ d, int n) {
    int i = blockIdx.x * blockDim.x + threadIdx.x;
    if (i < n) d[i] = (_Float16)s[i];
}

// cand = z[:,512:] * sigmoid(z[:,:512])   z: [M,1024] f32 -> cand [M,512] f32
__global__ void k_glu(const float* __restrict__ z, float* __restrict__ cand) {
    int idx = blockIdx.x * blockDim.x + threadIdx.x;
    if (idx >= MROWS * DMODEL) return;
    int c = idx & (DMODEL - 1);
    int m = idx >> 9;
    float gate = z[(size_t)m * 1024 + c];
    float cd   = z[(size_t)m * 1024 + DMODEL + c];
    cand[idx] = cd * (1.0f / (1.0f + __expf(-gate)));
}

// depthwise conv k=3 pad=1 along S (per batch) + bias + relu, f32 in -> f16 out
__global__ void k_conv_relu(const float* __restrict__ cand, const float* __restrict__ cw,
                            const float* __restrict__ cb, _Float16* __restrict__ out) {
    int idx = blockIdx.x * blockDim.x + threadIdx.x;
    if (idx >= MROWS * DMODEL) return;
    int c = idx & (DMODEL - 1);
    int m = idx >> 9;
    int s = m & (SEQ - 1);
    float left  = (s > 0)       ? cand[idx - DMODEL] : 0.f;
    float mid   = cand[idx];
    float right = (s < SEQ - 1) ? cand[idx + DMODEL] : 0.f;
    float v = left * cw[c * 3 + 0] + mid * cw[c * 3 + 1] + right * cw[c * 3 + 2] + cb[c];
    out[idx] = (_Float16)fmaxf(v, 0.f);
}

// LayerNorm: x f32 [M,512] -> y f16 [M,512]
__global__ __launch_bounds__(256) void k_ln(const float* __restrict__ x,
                                            const float* __restrict__ w,
                                            const float* __restrict__ b,
                                            _Float16* __restrict__ y) {
    const int D = DMODEL;
    int row = blockIdx.x;
    const float* xr = x + (size_t)row * D;
    float s = 0.f, ss = 0.f;
    for (int i = threadIdx.x; i < D; i += 256) { float v = xr[i]; s += v; ss += v * v; }
    __shared__ float sh1[256], sh2[256];
    sh1[threadIdx.x] = s; sh2[threadIdx.x] = ss;
    __syncthreads();
    for (int off = 128; off > 0; off >>= 1) {
        if (threadIdx.x < off) {
            sh1[threadIdx.x] += sh1[threadIdx.x + off];
            sh2[threadIdx.x] += sh2[threadIdx.x + off];
        }
        __syncthreads();
    }
    float mean = sh1[0] / D;
    float var  = sh2[0] / D - mean * mean;
    float rstd = rsqrtf(var + 1e-5f);
    for (int i = threadIdx.x; i < D; i += 256)
        y[(size_t)row * D + i] = (_Float16)((xr[i] - mean) * rstd * w[i] + b[i]);
}

// ------------------------------ WMMA GEMM ----------------------------------
// C[M,N] = epi(A[M,K] @ W[K,N] + bias)   A,W f16 row-major, f32 accumulate.
// Block = 256 threads (8 waves, wave32). 128x128 tile, double-buffered LDS.
// Wave grid 2(rows) x 4(cols): each wave 64 rows x 32 cols
//   -> 4 A-fragments x 2 B-fragments = 8 WMMAs per K-step, 64 accum VGPRs.
// EPI: 0 -> f32 out; 1 -> f16 out; 2 -> f32 out += residual; 3 -> f16 gelu.

template <int EPI>
__global__ __launch_bounds__(256) void k_gemm_wmma(const _Float16* __restrict__ A,
                                                   const _Float16* __restrict__ W,
                                                   const float* __restrict__ bias,
                                                   const float* __restrict__ R,
                                                   void* __restrict__ Cout,
                                                   int M, int N, int K) {
    __shared__ _Float16 As[2][128 * 32];   // [row][k], 64B rows, double buffered
    __shared__ _Float16 Bs[2][128 * 32];   // transposed: [n][k], 64B rows
    const int tid  = threadIdx.x;
    const int l    = tid & 31;             // lane (wave32)
    const int wv   = tid >> 5;             // wave 0..7
    const int wr   = wv >> 2;              // wave row 0..1  -> 64 rows each
    const int wc   = wv & 3;               // wave col 0..3  -> 32 cols each
    const int lm   = l & 15;
    const int lh   = l >> 4;
    const int row0 = blockIdx.y * 128;
    const int col0 = blockIdx.x * 128;

    // ---- per-thread staging addresses (advance by fixed strides per tile)
    const int sr = tid >> 2, sseg = tid & 3;           // A: 128 rows x 4 x 16B
    const _Float16* aSrc0 = A + (size_t)(row0 + sr) * K + sseg * 8;
    const _Float16* aSrc1 = aSrc0 + (size_t)64 * K;    // rows 64..127
    const unsigned  aOff0 = (unsigned)(sr * 64 + sseg * 16);
    const unsigned  aOff1 = aOff0 + 64 * 64;
    const int bk = tid >> 4;                           // W: k 0..15 (+16)
    const int bn = (tid & 15) * 8;                     //    n group of 8
    const _Float16* bSrc0 = W + (size_t)bk * N + col0 + bn;
    const _Float16* bSrc1 = bSrc0 + (size_t)16 * N;
    const size_t bStep = (size_t)32 * N;

    // stage tile into LDS buffer b: async A copy + register-transposed W tile
    auto stage = [&](int b) {
        unsigned dstBase = (unsigned)(uintptr_t)(void*)&As[b][0];
        asm volatile("global_load_async_to_lds_b128 %0, %1, off"
                     :: "v"(dstBase + aOff0), "v"((const void*)aSrc0) : "memory");
        asm volatile("global_load_async_to_lds_b128 %0, %1, off"
                     :: "v"(dstBase + aOff1), "v"((const void*)aSrc1) : "memory");
        v8h w0 = *(const v8h*)bSrc0;
        v8h w1 = *(const v8h*)bSrc1;
        _Float16* bDst = &Bs[b][bn * 32 + bk];
#pragma unroll
        for (int j = 0; j < 8; j++) bDst[j * 32] = w0[j];
#pragma unroll
        for (int j = 0; j < 8; j++) bDst[j * 32 + 16] = w1[j];
        aSrc0 += 32; aSrc1 += 32; bSrc0 += bStep; bSrc1 += bStep;
    };

    v8f c[4][2];
#pragma unroll
    for (int f = 0; f < 4; f++)
#pragma unroll
        for (int t = 0; t < 2; t++)
#pragma unroll
            for (int e = 0; e < 8; e++) c[f][t][e] = 0.f;

    const int nt = K / 32;
    stage(0);
    asm volatile("s_wait_asynccnt 0x0" ::: "memory");
    __syncthreads();

    for (int i = 0; i < nt; i++) {
        const int buf = i & 1;
        if (i + 1 < nt) {
            stage(buf ^ 1);                       // DMA of tile i+1 under compute
            __builtin_prefetch(bSrc0, 0, 3);      // pull tile i+2 W slice into L2
        }

        // ---- fragments + WMMA (tile i)
        const _Float16* asb = &As[buf][0];
        const _Float16* bsb = &Bs[buf][0];
        // B fragment (32x16): lane l -> n = lm; k = lh*16 + e
        v16h bf[2];
#pragma unroll
        for (int t = 0; t < 2; t++) {
            const v8h* br = (const v8h*)(bsb + (wc * 32 + t * 16 + lm) * 32);
            v8h blo = br[lh * 2];
            v8h bhi = br[lh * 2 + 1];
            bf[t] = __builtin_shufflevector(blo, bhi, 0, 1, 2, 3, 4, 5, 6, 7,
                                            8, 9, 10, 11, 12, 13, 14, 15);
        }
        // A fragment (16x32): lane l -> m = lm; K-half by lh
#pragma unroll
        for (int f = 0; f < 4; f++) {
            const v8h* ar = (const v8h*)(asb + (wr * 64 + f * 16 + lm) * 32);
            v8h alo = ar[lh * 2];
            v8h ahi = ar[lh * 2 + 1];
            v16h a = __builtin_shufflevector(alo, ahi, 0, 1, 2, 3, 4, 5, 6, 7,
                                             8, 9, 10, 11, 12, 13, 14, 15);
            c[f][0] = __builtin_amdgcn_wmma_f32_16x16x32_f16(
                false, a, false, bf[0], (short)0, c[f][0], false, false);
            c[f][1] = __builtin_amdgcn_wmma_f32_16x16x32_f16(
                false, a, false, bf[1], (short)0, c[f][1], false, false);
        }

        // publish tile i+1 (async A + ds-stored B) before anyone reads it
        asm volatile("s_wait_asynccnt 0x0" ::: "memory");
        __syncthreads();
    }

    // epilogue: c[f][t] reg r, lane l -> m = f*16 + lh*8 + r, n = t*16 + lm
#pragma unroll
    for (int f = 0; f < 4; f++) {
#pragma unroll
        for (int t = 0; t < 2; t++) {
            int n = col0 + wc * 32 + t * 16 + lm;
            float bv = bias[n];
#pragma unroll
            for (int r = 0; r < 8; r++) {
                int mrow = row0 + wr * 64 + f * 16 + lh * 8 + r;
                size_t off = (size_t)mrow * N + n;
                float v = c[f][t][r] + bv;
                if (EPI == 0) {
                    ((float*)Cout)[off] = v;
                } else if (EPI == 1) {
                    ((_Float16*)Cout)[off] = (_Float16)v;
                } else if (EPI == 2) {
                    ((float*)Cout)[off] = v + R[off];
                } else {
                    float g = 0.5f * v * (1.f + erff(v * 0.70710678118654752f));
                    ((_Float16*)Cout)[off] = (_Float16)g;
                }
            }
        }
    }
}

// --------------------------- flash attention -------------------------------
// One query per thread; K/V tiles (64 keys x 64 dims, f32) staged in LDS with
// RoPE applied on load. Online softmax; mask is all-true in this model.

__global__ __launch_bounds__(256) void k_flash_attn(const _Float16* __restrict__ Q,
                                                    const _Float16* __restrict__ K,
                                                    const _Float16* __restrict__ V,
                                                    _Float16* __restrict__ O) {
    const float LN1E4_OVER32 = 9.210340371976184f / 32.0f;  // ln(10000)/32
    int bh = blockIdx.y;
    int b = bh / NH, h = bh % NH;
    int s = blockIdx.x * 256 + threadIdx.x;
    size_t qoff = ((size_t)(b * SEQ + s)) * DMODEL + h * HDIM;

    float q[HDIM];
#pragma unroll
    for (int i = 0; i < HDIM; i++) q[i] = (float)Q[qoff + i];
#pragma unroll
    for (int i = 0; i < 32; i++) {       // RoPE on q
        float inv = __expf(-(float)i * LN1E4_OVER32);
        float sn, cs;
        __sincosf((float)s * inv, &sn, &cs);
        float a = q[i], bb = q[i + 32];
        q[i]      = a * cs - bb * sn;
        q[i + 32] = bb * cs + a * sn;
    }

    float mmax = -3.0e38f, lsum = 0.f;
    float acc[HDIM];
#pragma unroll
    for (int i = 0; i < HDIM; i++) acc[i] = 0.f;

    __shared__ float Ks[64][HDIM];
    __shared__ float Vs[64][HDIM];

    for (int kb = 0; kb < SEQ; kb += 64) {
        __syncthreads();
        for (int idx = threadIdx.x; idx < 64 * 32; idx += 256) {
            int j = idx >> 5, i = idx & 31;
            int pos = kb + j;
            size_t off = ((size_t)(b * SEQ + pos)) * DMODEL + h * HDIM;
            float k1 = (float)K[off + i], k2 = (float)K[off + i + 32];
            float inv = __expf(-(float)i * LN1E4_OVER32);
            float sn, cs;
            __sincosf((float)pos * inv, &sn, &cs);
            Ks[j][i]      = k1 * cs - k2 * sn;
            Ks[j][i + 32] = k2 * cs + k1 * sn;
            Vs[j][i]      = (float)V[off + i];
            Vs[j][i + 32] = (float)V[off + i + 32];
        }
        __syncthreads();
        for (int j = 0; j < 64; j++) {
            float dot = 0.f;
#pragma unroll
            for (int i = 0; i < HDIM; i++) dot += q[i] * Ks[j][i];
            dot *= 0.125f;  // 1/sqrt(64)
            float mn = fmaxf(mmax, dot);
            float corr = __expf(mmax - mn);
            float p = __expf(dot - mn);
            lsum = lsum * corr + p;
#pragma unroll
            for (int i = 0; i < HDIM; i++) acc[i] = acc[i] * corr + p * Vs[j][i];
            mmax = mn;
        }
    }
    float inv_l = 1.f / lsum;
#pragma unroll
    for (int i = 0; i < HDIM; i++) O[qoff + i] = (_Float16)(acc[i] * inv_l);
}

// ------------------------------ host side ----------------------------------

static void cvt(const float* s, _Float16* d, int n, hipStream_t st) {
    k_f32_to_f16<<<(n + 255) / 256, 256, 0, st>>>(s, d, n);
}

static void gemm(int epi, const _Float16* A, const _Float16* W, const float* bias,
                 const float* R, void* C, int N, int K, hipStream_t st) {
    dim3 g(N / 128, MROWS / 128), blk(256);
    switch (epi) {
        case 0: k_gemm_wmma<0><<<g, blk, 0, st>>>(A, W, bias, R, C, MROWS, N, K); break;
        case 1: k_gemm_wmma<1><<<g, blk, 0, st>>>(A, W, bias, R, C, MROWS, N, K); break;
        case 2: k_gemm_wmma<2><<<g, blk, 0, st>>>(A, W, bias, R, C, MROWS, N, K); break;
        default: k_gemm_wmma<3><<<g, blk, 0, st>>>(A, W, bias, R, C, MROWS, N, K); break;
    }
}

extern "C" void kernel_launch(void* const* d_in, const int* in_sizes, int n_in,
                              void* d_out, int out_size, void* d_ws, size_t ws_size,
                              hipStream_t stream) {
    (void)in_sizes; (void)n_in; (void)out_size; (void)ws_size;

    const float* frames     = (const float*)d_in[0];
    // d_in[1] = padding_mask (all true in this model) -- unused
    const float* input_w    = (const float*)d_in[2];
    const float* input_b    = (const float*)d_in[3];
    const float* ssm_norm_w = (const float*)d_in[4];
    const float* ssm_norm_b = (const float*)d_in[5];
    const float* ssm_in_w   = (const float*)d_in[6];
    const float* ssm_in_b   = (const float*)d_in[7];
    const float* ssm_conv_w = (const float*)d_in[8];
    const float* ssm_conv_b = (const float*)d_in[9];
    const float* ssm_out_w  = (const float*)d_in[10];
    const float* ssm_out_b  = (const float*)d_in[11];
    const float* a_ln1_w    = (const float*)d_in[12];
    const float* a_ln1_b    = (const float*)d_in[13];
    const float* a_q_w      = (const float*)d_in[14];
    const float* a_q_b      = (const float*)d_in[15];
    const float* a_k_w      = (const float*)d_in[16];
    const float* a_k_b      = (const float*)d_in[17];
    const float* a_v_w      = (const float*)d_in[18];
    const float* a_v_b      = (const float*)d_in[19];
    const float* a_o_w      = (const float*)d_in[20];
    const float* a_o_b      = (const float*)d_in[21];
    const float* a_ln2_w    = (const float*)d_in[22];
    const float* a_ln2_b    = (const float*)d_in[23];
    const float* a_f1_w     = (const float*)d_in[24];
    const float* a_f1_b     = (const float*)d_in[25];
    const float* a_f2_w     = (const float*)d_in[26];
    const float* a_f2_b     = (const float*)d_in[27];
    const float* fn_w       = (const float*)d_in[28];
    const float* fn_b       = (const float*)d_in[29];
    const float* frame_w    = (const float*)d_in[30];
    const float* frame_b    = (const float*)d_in[31];
    const float* sym_w      = (const float*)d_in[32];
    const float* sym_b      = (const float*)d_in[33];

    // workspace layout
    char* ws = (char*)d_ws;
    float*     X   = (float*)ws;                               // 16 MB  residual stream f32
    _Float16*  A16 = (_Float16*)(ws + (16ull << 20));          //  8 MB  f16 GEMM-input slot
    _Float16*  B16 = (_Float16*)(ws + (24ull << 20));          // 32 MB  q|k|v / ff intermediate
    float*     Z   = (float*)(ws + (56ull << 20));             // 32 MB  ssm in-proj out f32
    float*     C32 = (float*)(ws + (88ull << 20));             // 16 MB  glu cand f32
    _Float16*  W16 = (_Float16*)(ws + (104ull << 20));         //  2 MB  weight f16 scratch
    _Float16* Q16 = B16;
    _Float16* K16 = B16 + (size_t)MROWS * DMODEL;
    _Float16* V16 = B16 + (size_t)MROWS * DMODEL * 2;

    const int MD = MROWS * DMODEL;
    dim3 ew((MD + 255) / 256), ewb(256);

    // input projection: x = frames @ input_w + input_b
    cvt(frames, A16, MD, stream);
    cvt(input_w, W16, DMODEL * DMODEL, stream);
    gemm(0, A16, W16, input_b, nullptr, X, DMODEL, DMODEL, stream);

    int si = 0, ai = 0;
    for (int layer = 0; layer < 12; layer++) {
        if (layer == 4 || layer == 8) {
            // ---- attention block ----
            k_ln<<<MROWS, 256, 0, stream>>>(X, a_ln1_w + ai * DMODEL, a_ln1_b + ai * DMODEL, A16);
            cvt(a_q_w + (size_t)ai * DMODEL * DMODEL, W16, DMODEL * DMODEL, stream);
            gemm(1, A16, W16, a_q_b + ai * DMODEL, nullptr, Q16, DMODEL, DMODEL, stream);
            cvt(a_k_w + (size_t)ai * DMODEL * DMODEL, W16, DMODEL * DMODEL, stream);
            gemm(1, A16, W16, a_k_b + ai * DMODEL, nullptr, K16, DMODEL, DMODEL, stream);
            cvt(a_v_w + (size_t)ai * DMODEL * DMODEL, W16, DMODEL * DMODEL, stream);
            gemm(1, A16, W16, a_v_b + ai * DMODEL, nullptr, V16, DMODEL, DMODEL, stream);

            dim3 fg(SEQ / 256, NB * NH);
            k_flash_attn<<<fg, 256, 0, stream>>>(Q16, K16, V16, A16);

            cvt(a_o_w + (size_t)ai * DMODEL * DMODEL, W16, DMODEL * DMODEL, stream);
            gemm(2, A16, W16, a_o_b + ai * DMODEL, X, X, DMODEL, DMODEL, stream);

            k_ln<<<MROWS, 256, 0, stream>>>(X, a_ln2_w + ai * DMODEL, a_ln2_b + ai * DMODEL, A16);
            cvt(a_f1_w + (size_t)ai * DMODEL * 2048, W16, DMODEL * 2048, stream);
            gemm(3, A16, W16, a_f1_b + ai * 2048, nullptr, B16, 2048, DMODEL, stream);
            cvt(a_f2_w + (size_t)ai * 2048 * DMODEL, W16, 2048 * DMODEL, stream);
            gemm(2, B16, W16, a_f2_b + ai * DMODEL, X, X, DMODEL, 2048, stream);
            ai++;
        } else {
            // ---- SSM block ----
            k_ln<<<MROWS, 256, 0, stream>>>(X, ssm_norm_w + si * DMODEL, ssm_norm_b + si * DMODEL, A16);
            cvt(ssm_in_w + (size_t)si * DMODEL * 1024, W16, DMODEL * 1024, stream);
            gemm(0, A16, W16, ssm_in_b + si * 1024, nullptr, Z, 1024, DMODEL, stream);
            k_glu<<<ew, ewb, 0, stream>>>(Z, C32);
            k_conv_relu<<<ew, ewb, 0, stream>>>(C32, ssm_conv_w + si * DMODEL * 3,
                                                ssm_conv_b + si * DMODEL, A16);
            cvt(ssm_out_w + (size_t)si * DMODEL * DMODEL, W16, DMODEL * DMODEL, stream);
            gemm(2, A16, W16, ssm_out_b + si * DMODEL, X, X, DMODEL, DMODEL, stream);
            si++;
        }
    }

    // final norm + heads
    k_ln<<<MROWS, 256, 0, stream>>>(X, fn_w, fn_b, A16);
    float* out0 = (float*)d_out;                          // [B,S,F]
    float* out1 = (float*)d_out + (size_t)MROWS * DMODEL; // [B,S,V]
    cvt(frame_w, W16, DMODEL * DMODEL, stream);
    gemm(0, A16, W16, frame_b, nullptr, out0, DMODEL, DMODEL, stream);
    cvt(sym_w, W16, DMODEL * 1024, stream);
    gemm(0, A16, W16, sym_b, nullptr, out1, 1024, DMODEL, stream);
}
